// EmbeddingSharedWeights_8546984919124
// MI455X (gfx1250) — compile-verified, matrix operand order checked by reading, not run on hardware
//
#include <hip/hip_runtime.h>

// EmbeddingSharedWeights: out[m, :] = (w0[tok[m], :] @ w1) * 32 * (tok[m] != 0)
//   M = 16384 tokens, K = 2048, N = 1024, all f32 in memory.
// Strategy: gathered GEMM with bf16 WMMA (f32 accumulation), LDS double buffer.

#define HIDDEN   1024
#define TWOH     2048
#define M_TOKENS 16384
#define BM 128
#define BN 128
#define BK 32
#define KTILES (TWOH / BK)   // 64
#define AK (BK + 8)          // padded LDS row pitch in halves (80 bytes)
#define THREADS 256

typedef __attribute__((ext_vector_type(16))) __bf16 v16bf;
typedef __attribute__((ext_vector_type(8)))  __bf16 v8bf;
typedef __attribute__((ext_vector_type(8)))  float   v8f;
typedef __attribute__((ext_vector_type(4)))  float   v4f;

__global__ __launch_bounds__(THREADS)
void emb_wmma_kernel(const int* __restrict__ tokens,
                     const float* __restrict__ w0,
                     const float* __restrict__ w1,
                     float* __restrict__ out)
{
    __shared__ __align__(16) __bf16 As[2][BM][AK];   // gathered w0 tile (bf16)
    __shared__ __align__(16) __bf16 Bs[2][BN][AK];   // w1 tile, transposed [n][k]
    __shared__ int   tokS[BM];
    __shared__ float scaleS[BM];

    const int t      = threadIdx.x;
    const int mBlock = blockIdx.x * BM;
    const int nBlock = blockIdx.y * BN;

    if (t < BM) {
        const int tok = tokens[mBlock + t];
        tokS[t]   = tok;
        scaleS[t] = (tok != 0) ? 32.0f : 0.0f;   // sqrt(HIDDEN) = 32 exactly
    }
    __syncthreads();

    // ---- cooperative-load coordinates (fixed per thread) ----
    const int a_kq = t & 7;    // which float4 along K (8 per 32-wide row)
    const int a_m0 = t >> 3;   // first of 4 rows (stride 32)
    const int b_nq = t & 31;   // which float4 along N (32 per 128-wide row)
    const int b_k0 = t >> 5;   // first of 4 k-rows (stride 8)

    // Gather row bases once: every K iteration reuses the same 128 rows.
    size_t aRowOff[4];
    #pragma unroll
    for (int i = 0; i < 4; ++i)
        aRowOff[i] = (size_t)tokS[a_m0 + i * 32] * (size_t)TWOH;

    v4f aReg[4], bReg[4];

    auto loadGlobal = [&](int kt) {
        const int k0 = kt * BK;
        #pragma unroll
        for (int i = 0; i < 4; ++i)
            aReg[i] = *(const v4f*)(w0 + aRowOff[i] + (k0 + a_kq * 4));
        #pragma unroll
        for (int i = 0; i < 4; ++i)
            bReg[i] = *(const v4f*)(w1 + (size_t)(k0 + b_k0 + i * 8) * HIDDEN
                                        + nBlock + b_nq * 4);
    };

    auto storeSmem = [&](int buf) {
        #pragma unroll
        for (int i = 0; i < 4; ++i) {
            __bf16* d = &As[buf][a_m0 + i * 32][a_kq * 4];
            #pragma unroll
            for (int j = 0; j < 4; ++j) d[j] = (__bf16)aReg[i][j];
        }
        #pragma unroll
        for (int i = 0; i < 4; ++i) {
            const int kr = b_k0 + i * 8;
            #pragma unroll
            for (int j = 0; j < 4; ++j)
                Bs[buf][b_nq * 4 + j][kr] = (__bf16)bReg[i][j];   // transpose
        }
    };

    // ---- wave/lane mapping for WMMA compute ----
    const int wave = t >> 5;        // 8 waves
    const int lane = t & 31;
    const int wm   = wave >> 1;     // 0..3 -> 32 rows each
    const int wn   = wave & 1;      // 0..1 -> 64 cols each
    const int r    = lane & 15;
    const int half = lane >> 4;

    v8f acc[2][4] = {};

    loadGlobal(0);
    storeSmem(0);
    __syncthreads();

    for (int kt = 0; kt < KTILES; ++kt) {
        const int cur = kt & 1;
        if (kt + 1 < KTILES) loadGlobal(kt + 1);

        // Fragment loads from LDS.  16-bit A layout (ISA 7.12.2):
        //   lane half 0: K = 0..7 (V0..V3) and 16..23 (V4..V7)
        //   lane half 1: K = 8..15         and 24..31
        // B mirrors A with column = lane&15 (Bs stored as [n][k]).
        v16bf af[2], bfr[4];
        #pragma unroll
        for (int i = 0; i < 2; ++i) {
            const __bf16* p = &As[cur][wm * 32 + i * 16 + r][half * 8];
            v8bf lo = *(const v8bf*)p;          // K = half*8 .. +7
            v8bf hi = *(const v8bf*)(p + 16);   // K = 16 + half*8 .. +7
            af[i] = __builtin_shufflevector(lo, hi,
                     0,1,2,3,4,5,6,7,8,9,10,11,12,13,14,15);
        }
        #pragma unroll
        for (int j = 0; j < 4; ++j) {
            const __bf16* p = &Bs[cur][wn * 64 + j * 16 + r][half * 8];
            v8bf lo = *(const v8bf*)p;
            v8bf hi = *(const v8bf*)(p + 16);
            bfr[j] = __builtin_shufflevector(lo, hi,
                     0,1,2,3,4,5,6,7,8,9,10,11,12,13,14,15);
        }

        #pragma unroll
        for (int i = 0; i < 2; ++i)
            #pragma unroll
            for (int j = 0; j < 4; ++j)
                acc[i][j] = __builtin_amdgcn_wmma_f32_16x16x32_bf16(
                    false, af[i], false, bfr[j], (short)0, acc[i][j],
                    false, false);

        if (kt + 1 < KTILES) storeSmem(cur ^ 1);
        __syncthreads();
    }

    // ---- epilogue: mask + scale + store ----
    // C/D layout: VGPR v holds row (half*8 + v) of the 16x16 tile, col = lane&15.
    #pragma unroll
    for (int i = 0; i < 2; ++i) {
        const int rlBase = wm * 32 + i * 16 + half * 8;
        #pragma unroll
        for (int j = 0; j < 4; ++j) {
            const int col = nBlock + wn * 64 + j * 16 + r;
            #pragma unroll
            for (int v = 0; v < 8; ++v) {
                const int rl = rlBase + v;
                out[(size_t)(mBlock + rl) * HIDDEN + col] =
                    acc[i][j][v] * scaleS[rl];
            }
        }
    }
}

extern "C" void kernel_launch(void* const* d_in, const int* in_sizes, int n_in,
                              void* d_out, int out_size, void* d_ws, size_t ws_size,
                              hipStream_t stream)
{
    (void)in_sizes; (void)n_in; (void)out_size; (void)d_ws; (void)ws_size;
    const int*   tokens = (const int*)d_in[0];   // [4,4096] int32
    const float* w0     = (const float*)d_in[1]; // [32000, 2048] f32
    const float* w1     = (const float*)d_in[2]; // [2048, 1024] f32
    float*       out    = (float*)d_out;         // [4,4096,1024] f32

    dim3 grid(M_TOKENS / BM, HIDDEN / BN);       // 128 x 8 blocks
    emb_wmma_kernel<<<grid, THREADS, 0, stream>>>(tokens, w0, w1, out);
}
